// FullyConnectedTensorProductRescaleSwishGate_28913719837199
// MI455X (gfx1250) — matrix operand order repaired and checked by reading, not compiled, not run on hardware
//
#include <hip/hip_runtime.h>
#include <math.h>

typedef __attribute__((ext_vector_type(16))) _Float16 v16h;
typedef __attribute__((ext_vector_type(8)))  _Float16 v8h;
typedef __attribute__((ext_vector_type(8)))  float    v8f;

#define MUL_S 256
#define MUL_V 128
#define DIM1  640

// feature columns (f16, per row)
#define FC_X0Y0 0
#define FC_DOT  256
#define FC_X0   384
#define FC_X1   640     // + comp*128
#define FC_CR   1024    // + comp*128
#define FSTR    1416    // padded row stride in halves (2832 B, 16B-aligned, bank-skewed)

// staging columns (f32, per row)
#define SC_S1   0
#define SC_S2   256
#define SC_P4   384
#define SC_Q5   512     // + comp*128
#define SC_Q6   896     // + comp*128
#define OSTR    1288

// x staging (f32, aliases stg region; used only before phase 2)
#define XSTR    644     // padded: 644 % 64 == 4 -> bank skew, float4-aligned

#define ROWS_PER_WG 32

// f16 weight offsets in d_ws (units: halves). All stored transposed: Wt[n][k].
#define OFF_W0 0        // 256x256
#define OFF_W1 65536    // 256x128
#define OFF_W2 98304    // 128x256
#define OFF_W3 131072   // 128x128
#define OFF_W4 147456   // 128x256
#define OFF_W5 180224   // 128x128
#define OFF_W6 196608   // 128x128

#define SMEM_BYTES (ROWS_PER_WG*FSTR*2 + ROWS_PER_WG*OSTR*4 + ROWS_PER_WG*4*4)  // 256000 B

// ---------------- weight conversion: f32 row-major (K,N) -> f16 transposed (N,K), scaled ----
__global__ void convert_w_kernel(const float* __restrict__ src, _Float16* __restrict__ dst,
                                 int K, int Nout, float scale) {
  int i = blockIdx.x * 256 + threadIdx.x;
  if (i < K * Nout) {
    int k = i / Nout;
    int nn = i - k * Nout;
    dst[(size_t)nn * K + k] = (_Float16)(scale * src[i]);
  }
}

// ---------------- one 16x16 output tile, K-loop of WMMA f16 ----------------
template<int K>
__device__ __forceinline__ void gemm_tile(v8f& acc,
                                          const _Float16* arow,   // LDS: row base + hf*16
                                          const _Float16* brow) { // global: Wt row base + hf*16
#pragma unroll
  for (int k = 0; k < K; k += 32) {
    v8h  alo = *(const v8h*)(arow + k);        // K = k + hf*16 .. +7
    v8h  ahi = *(const v8h*)(arow + k + 8);    // K = k + hf*16 + 8 .. +15
    v16h a   = __builtin_shufflevector(alo, ahi, 0,1,2,3,4,5,6,7,8,9,10,11,12,13,14,15);
    v16h b   = *(const v16h*)(brow + k);       // K = k + hf*16 .. +15, n = lid
    acc = __builtin_amdgcn_wmma_f32_16x16x32_f16(false, a, false, b, (short)0, acc,
                                                 false, false);
  }
}

// ---------------- fused tensor-product + gate kernel ----------------
__global__ __launch_bounds__(256, 1)
void fused_tp_kernel(const float* __restrict__ xg, const float* __restrict__ yg,
                     const _Float16* __restrict__ ws, const float* __restrict__ bias,
                     float* __restrict__ outg, int n,
                     float inv, float silu_c, float sig_c) {
  extern __shared__ __align__(16) char smem[];
  _Float16* feat = (_Float16*)smem;                         // 32 x FSTR halves
  float*    stg  = (float*)(smem + ROWS_PER_WG*FSTR*2);     // 32 x OSTR floats
  float*    xsh  = stg;                                     // aliases stg (phase 0/1 only)
  float*    ysh  = stg + ROWS_PER_WG*OSTR;                  // 32 x 4 floats

  const int tid  = threadIdx.x;
  const int row0 = blockIdx.x * ROWS_PER_WG;

  // ---- phase 0: coalesced staging of y and the 32x640 x block into LDS ----
  for (int i = tid; i < ROWS_PER_WG*4; i += 256) {
    int r = i >> 2;
    ysh[i] = (row0 + r < n) ? yg[(size_t)(row0 + r)*4 + (i & 3)] : 0.f;
  }
  for (int i = tid; i < ROWS_PER_WG*(DIM1/4); i += 256) {   // 160 float4 per row
    int r = i / (DIM1/4), q = i - r*(DIM1/4);
    float4 v = make_float4(0.f, 0.f, 0.f, 0.f);
    if (row0 + r < n)
      v = ((const float4*)(xg + (size_t)(row0 + r)*DIM1))[q];
    *(float4*)(xsh + (size_t)r*XSTR + q*4) = v;
  }
  if (row0 + ROWS_PER_WG + (tid >> 3) < n)                  // prefetch next block's x
    __builtin_prefetch(xg + (size_t)(row0 + ROWS_PER_WG + (tid >> 3))*DIM1 + (tid & 7)*80, 0, 1);
  __syncthreads();

  // ---- phase 1: build f16 feature matrix in LDS ----
  {
    int r = tid >> 3, t = tid & 7;          // 8 threads per row
    const float* xrow = xsh + (size_t)r*XSTR;
    float y0 = ysh[r*4+0], ya = ysh[r*4+1], yb = ysh[r*4+2], yc = ysh[r*4+3];
    _Float16* frow = feat + r * FSTR;
#pragma unroll 4
    for (int j = 0; j < 32; ++j) {
      int c = t*32 + j;
      float xv = xrow[c];
      frow[FC_X0Y0 + c] = (_Float16)(xv * y0);
      frow[FC_X0   + c] = (_Float16)xv;
    }
#pragma unroll 2
    for (int j = 0; j < 16; ++j) {
      int u = t*16 + j;
      float a0 = xrow[MUL_S + u*3 + 0];
      float a1 = xrow[MUL_S + u*3 + 1];
      float a2 = xrow[MUL_S + u*3 + 2];
      frow[FC_DOT + u]          = (_Float16)(a0*ya + a1*yb + a2*yc);
      frow[FC_X1 + 0*128 + u]   = (_Float16)a0;
      frow[FC_X1 + 1*128 + u]   = (_Float16)a1;
      frow[FC_X1 + 2*128 + u]   = (_Float16)a2;
      frow[FC_CR + 0*128 + u]   = (_Float16)(a1*yc - a2*yb);   // cross(x1, y1)
      frow[FC_CR + 1*128 + u]   = (_Float16)(a2*ya - a0*yc);
      frow[FC_CR + 2*128 + u]   = (_Float16)(a0*yb - a1*ya);
    }
  }
  __syncthreads();

  // ---- phase 2: WMMA GEMMs (scalar job decode; EXEC all-ones throughout) ----
  {
    const int wave_s = __builtin_amdgcn_readfirstlane(tid >> 5);  // SGPR wave id
    const int rb = wave_s & 1;               // which 16-row block
    const int wq = wave_s >> 1;              // job slice 0..3
    const int lane = tid & 31;
    const int lid = lane & 15, hf = lane >> 4;
    const _Float16* abase = feat + (rb*16 + lid) * FSTR + hf*16;

    for (int job = wq; job < 80; job += 4) { // scalar loop/branches
      v8f acc = {};
      int outc;
      if (job < 16) {                        // s1 tiles: F0@W0 + F1@W1  (N=256)
        int n16 = job;
        gemm_tile<256>(acc, abase + FC_X0Y0, ws + OFF_W0 + (n16*16 + lid)*256 + hf*16);
        gemm_tile<128>(acc, abase + FC_DOT,  ws + OFF_W1 + (n16*16 + lid)*128 + hf*16);
        outc = SC_S1 + n16*16;
      } else if (job < 24) {                 // s2 tiles: F0@W2 + F1@W3  (N=128)
        int n16 = job - 16;
        gemm_tile<256>(acc, abase + FC_X0Y0, ws + OFF_W2 + (n16*16 + lid)*256 + hf*16);
        gemm_tile<128>(acc, abase + FC_DOT,  ws + OFF_W3 + (n16*16 + lid)*128 + hf*16);
        outc = SC_S2 + n16*16;
      } else if (job < 32) {                 // p4 tiles: F2@W4          (N=128)
        int n16 = job - 24;
        gemm_tile<256>(acc, abase + FC_X0,   ws + OFF_W4 + (n16*16 + lid)*256 + hf*16);
        outc = SC_P4 + n16*16;
      } else if (job < 56) {                 // q5: x1_i@W5, 3 comps x 8 tiles
        int t = job - 32, comp = t >> 3, n16 = t & 7;
        gemm_tile<128>(acc, abase + FC_X1 + comp*128,
                       ws + OFF_W5 + (n16*16 + lid)*128 + hf*16);
        outc = SC_Q5 + comp*128 + n16*16;
      } else {                               // q6: cross_i@W6
        int t = job - 56, comp = t >> 3, n16 = t & 7;
        gemm_tile<128>(acc, abase + FC_CR + comp*128,
                       ws + OFF_W6 + (n16*16 + lid)*128 + hf*16);
        outc = SC_Q6 + comp*128 + n16*16;
      }
      // D layout: lane holds column n=lid; VGPR r holds row M = r + hf*8
      float* srow = stg + (size_t)(rb*16 + hf*8) * OSTR + outc + lid;
#pragma unroll
      for (int r2 = 0; r2 < 8; ++r2) srow[(size_t)r2 * OSTR] = acc[r2];
    }
  }
  __syncthreads();

  // ---- phase 3: epilogue ----
  // scalar outputs: SiLU, float4 vectorized coalesced stores
  for (int i = tid; i < ROWS_PER_WG*(MUL_S/4); i += 256) {  // 64 float4 per row
    int r = i >> 6, c4 = (i & 63) * 4;
    int grow = row0 + r;
    if (grow < n) {
      float4 o;
      float* po = &o.x;
#pragma unroll
      for (int e = 0; e < 4; ++e) {
        float v = stg[r*OSTR + SC_S1 + c4 + e] * inv + bias[c4 + e];
        float s = 1.f / (1.f + __expf(-v));
        po[e] = silu_c * v * s;
      }
      *(float4*)(outg + (size_t)grow*DIM1 + c4) = o;
    }
  }
  // vector outputs: sigmoid gate * (p4*y1 + q5*y0 + q6)
  for (int i = tid; i < ROWS_PER_WG*MUL_V; i += 256) {
    int r = i >> 7, w = i & 127;
    int grow = row0 + r;
    if (grow < n) {
      const float* srow = stg + r*OSTR;
      float g  = sig_c / (1.f + __expf(-(srow[SC_S2 + w]*inv + bias[MUL_S + w])));
      float y0 = ysh[r*4 + 0];
      float p4 = srow[SC_P4 + w];
      float* orow = outg + (size_t)grow*DIM1 + MUL_S + w*3;
#pragma unroll
      for (int comp = 0; comp < 3; ++comp) {
        float vv = p4 * ysh[r*4 + 1 + comp]
                 + srow[SC_Q5 + comp*128 + w] * y0
                 + srow[SC_Q6 + comp*128 + w];
        orow[comp] = vv * inv * g;
      }
    }
  }
}

extern "C" void kernel_launch(void* const* d_in, const int* in_sizes, int n_in,
                              void* d_out, int out_size, void* d_ws, size_t ws_size,
                              hipStream_t stream) {
  const float* x = (const float*)d_in[0];
  const float* y = (const float*)d_in[1];
  const float* W[7] = {(const float*)d_in[2], (const float*)d_in[3], (const float*)d_in[4],
                       (const float*)d_in[5], (const float*)d_in[6], (const float*)d_in[7],
                       (const float*)d_in[8]};
  const float* b = (const float*)d_in[9];
  float* out = (float*)d_out;
  _Float16* ws = (_Float16*)d_ws;
  const int n = in_sizes[0] / DIM1;

  // normalization constants (folded into converted weights)
  const float C0 = (float)sqrt(65536.0 / 384.0);
  const float C1 = (float)(sqrt(32768.0 / 384.0) / sqrt(3.0));
  const float C2 = (float)sqrt(32768.0 / 384.0);
  const float C3 = (float)(sqrt(16384.0 / 384.0) / sqrt(3.0));
  const float C4 = 8.0f;                       // sqrt(3*128*256/512)/sqrt(3)
  const float C5 = (float)sqrt(32.0);          // sqrt(3*128*128/512)/sqrt(3)
  const float C6 = 4.0f;                       // sqrt(3*128*128/512)/sqrt(6)

  struct Cfg { int K, N; size_t off; const float* src; float c; };
  const Cfg cfg[7] = {
    {256, 256, OFF_W0, W[0], C0},
    {128, 256, OFF_W1, W[1], C1},
    {256, 128, OFF_W2, W[2], C2},
    {128, 128, OFF_W3, W[3], C3},
    {256, 128, OFF_W4, W[4], C4},
    {128, 128, OFF_W5, W[5], C5},
    {128, 128, OFF_W6, W[6], C6},
  };
  for (int i = 0; i < 7; ++i) {
    int tot = cfg[i].K * cfg[i].N;
    convert_w_kernel<<<(tot + 255) / 256, 256, 0, stream>>>(
        cfg[i].src, ws + cfg[i].off, cfg[i].K, cfg[i].N, cfg[i].c);
  }

  static_assert(SMEM_BYTES <= 320 * 1024, "LDS over WGP limit");
  static_assert(ROWS_PER_WG * XSTR * 4 <= ROWS_PER_WG * OSTR * 4, "xstage must fit in stg");
  (void)hipFuncSetAttribute(reinterpret_cast<const void*>(fused_tp_kernel),
                            hipFuncAttributeMaxDynamicSharedMemorySize, SMEM_BYTES);

  const float inv    = (float)(1.0 / sqrt(2560.0));   // 1/sqrt(FAN_IN)
  const float SILU_C = 1.6765312f;   // 1/sqrt(E[silu(x)^2]), x~N(0,1)
  const float SIG_C  = 1.8466852f;   // 1/sqrt(E[sigmoid(x)^2])

  const int grid = (n + ROWS_PER_WG - 1) / ROWS_PER_WG;
  fused_tp_kernel<<<grid, 256, SMEM_BYTES, stream>>>(x, y, ws, b, out, n,
                                                     inv, SILU_C, SIG_C);
}
